// GAttention_62122406969868
// MI455X (gfx1250) — matrix operand order repaired and checked by reading, not compiled, not run on hardware
//
#include <hip/hip_runtime.h>
#include <hip/hip_bf16.h>

// ---------------------------------------------------------------------------
// GAttention for MI455X (gfx1250, wave32, WMMA).
//   1) cvt_kernel      : x fp32 -> bf16
//   2) cvt_t_kernel    : W_qk, W_v fp32 [K][N] -> bf16 [N][K]
//   3) gemm_bf16_kernel<T>: qk = x@Wqk+b (row-major bf16)
//                           v  = x@Wv +b (transposed vT[b*1024+n][s], packed b128 stores)
//   4) mass_sq_kernel  : mass = softplus(x@Wm+b); sq = |qk_row|^2   (fp32 [bh][s])
//   5) attn_kernel     : flash-style online softmax; qk_j / v_j tiles double-buffered
//                        in LDS via global_load_async_to_lds_b128 (ASYNCcnt), so the
//                        async DMA for step j+1 overlaps the WMMAs of step j.
// ---------------------------------------------------------------------------

#define EMB   1024
#define HEADS 16
#define HDIM  64
#define BSZ   2
#define SEQ   2048
#define ROWS  (BSZ * SEQ)          // 4096

typedef __bf16 bf16;
typedef __attribute__((ext_vector_type(16))) __bf16 v16bf;
typedef __attribute__((ext_vector_type(8)))  __bf16 v8bf;
typedef __attribute__((ext_vector_type(8)))  float  v8f;

#define WMMA_BF16(A, B, C) \
  __builtin_amdgcn_wmma_f32_16x16x32_bf16(false, (A), false, (B), (short)0, (C), false, false)

#if defined(__gfx1250__) && __has_builtin(__builtin_amdgcn_global_load_async_to_lds_b128)
#define HAVE_ASYNC_LDS 1
#else
#define HAVE_ASYNC_LDS 0
#endif

#if HAVE_ASYNC_LDS
typedef int v4i_ __attribute__((vector_size(16)));
typedef __attribute__((address_space(1))) v4i_* as1_v4i;
typedef __attribute__((address_space(3))) v4i_* as3_v4i;
#endif

union FragU { v16bf v; v8bf h[2]; };

// A-fragment: lane holds K {base..base+7} then {base+16..base+23} (two 16B chunks)
__device__ __forceinline__ v16bf load_frag_pair(const bf16* p_lo, const bf16* p_hi) {
  FragU f;
  f.h[0] = *reinterpret_cast<const v8bf*>(p_lo);
  f.h[1] = *reinterpret_cast<const v8bf*>(p_hi);
  return f.v;
}
// B-fragment: lane holds K {base..base+7} then {base+16..base+23} (two 16B chunks)
__device__ __forceinline__ v16bf load_frag_c(const bf16* p) {
  return load_frag_pair(p, p + 16);
}

// 16B global -> LDS staging: async-to-LDS (ASYNCcnt) when available, else copy.
__device__ __forceinline__ void stage16(const bf16* g, bf16* l) {
#if HAVE_ASYNC_LDS
  __builtin_amdgcn_global_load_async_to_lds_b128(
      (as1_v4i)(__attribute__((address_space(1))) const void*)g,
      (as3_v4i)(__attribute__((address_space(3))) void*)l,
      0, 0);
#else
  *reinterpret_cast<v8bf*>(l) = *reinterpret_cast<const v8bf*>(g);
#endif
}

__device__ __forceinline__ void stage_wait() {
#if HAVE_ASYNC_LDS
#if __has_builtin(__builtin_amdgcn_s_wait_asynccnt)
  __builtin_amdgcn_s_wait_asynccnt(0);
#else
  asm volatile("s_wait_asynccnt 0" ::: "memory");
#endif
#endif
}

// ---------------------------------------------------------------------------
__global__ void __launch_bounds__(256) cvt_kernel(const float* __restrict__ src,
                                                  bf16* __restrict__ dst, int n) {
  int i = blockIdx.x * 256 + threadIdx.x;
  if (i < n) dst[i] = (bf16)src[i];
}

__global__ void __launch_bounds__(256) cvt_t_kernel(const float* __restrict__ src,
                                                    bf16* __restrict__ dst) {
  int i = blockIdx.x * 256 + threadIdx.x;      // i over 1024*1024
  int n = i >> 10;
  int k = i & 1023;
  dst[(size_t)n * 1024 + k] = (bf16)src[(size_t)k * 1024 + n];
}

// ---------------------------------------------------------------------------
// C[4096,1024] = A[4096,1024] @ B (given as B^T [N][K]) + bias.
// 8 waves (4M x 2N); wave tile 32M x 64N; fragments direct from global (L2-hot).
template <int TRANSPOSE>
__global__ void __launch_bounds__(256)
gemm_bf16_kernel(const bf16* __restrict__ A, const bf16* __restrict__ BT,
                 const float* __restrict__ bias, bf16* __restrict__ C) {
  const int wid  = threadIdx.x >> 5;
  const int lane = threadIdx.x & 31;
  const int l15  = lane & 15;
  const int hi   = lane >> 4;
  const int wm   = wid & 3;
  const int wn   = wid >> 2;
  const int m0   = blockIdx.x * 128 + wm * 32;
  const int n0   = blockIdx.y * 128 + wn * 64;

  v8f acc[2][4];
#pragma unroll
  for (int mt = 0; mt < 2; ++mt)
#pragma unroll
    for (int nt = 0; nt < 4; ++nt)
      acc[mt][nt] = (v8f){0.f, 0.f, 0.f, 0.f, 0.f, 0.f, 0.f, 0.f};

  const int akb = hi * 8;
  for (int k0 = 0; k0 < 1024; k0 += 32) {
    v16bf af[2], bfr[4];
#pragma unroll
    for (int mt = 0; mt < 2; ++mt) {
      const bf16* p = A + (size_t)(m0 + mt * 16 + l15) * 1024 + k0 + akb;
      af[mt] = load_frag_pair(p, p + 16);
      __builtin_prefetch(p + 32, 0, 3);
    }
#pragma unroll
    for (int nt = 0; nt < 4; ++nt) {
      const bf16* p = BT + (size_t)(n0 + nt * 16 + l15) * 1024 + k0 + hi * 8;
      bfr[nt] = load_frag_c(p);
    }
#pragma unroll
    for (int mt = 0; mt < 2; ++mt)
#pragma unroll
      for (int nt = 0; nt < 4; ++nt)
        acc[mt][nt] = WMMA_BF16(af[mt], bfr[nt], acc[mt][nt]);
  }

#pragma unroll
  for (int nt = 0; nt < 4; ++nt) {
    const int   col = n0 + nt * 16 + l15;
    const float bv  = bias[col];
#pragma unroll
    for (int mt = 0; mt < 2; ++mt) {
      if (TRANSPOSE) {
        // rows r=0..7 -> consecutive s: one packed 16B store per tile per lane
        const int row0 = m0 + mt * 16 + hi * 8;
        const int b    = row0 >> 11, s0 = row0 & 2047;
        v8bf pack;
#pragma unroll
        for (int r = 0; r < 8; ++r) pack[r] = (bf16)(acc[mt][nt][r] + bv);
        *reinterpret_cast<v8bf*>(&C[((size_t)(b * 1024 + col)) * SEQ + s0]) = pack;
      } else {
#pragma unroll
        for (int r = 0; r < 8; ++r) {
          const int row = m0 + mt * 16 + r + hi * 8;
          C[(size_t)row * 1024 + col] = (bf16)(acc[mt][nt][r] + bv);
        }
      }
    }
  }
}

// ---------------------------------------------------------------------------
__global__ void __launch_bounds__(256)
mass_sq_kernel(const float* __restrict__ x, const float* __restrict__ Wm,
               const float* __restrict__ bm, const bf16* __restrict__ qkb,
               float* __restrict__ mass, float* __restrict__ sq) {
  const int t   = blockIdx.x * 256 + threadIdx.x;  // 65536 total
  const int row = t >> 4;
  const int h   = t & 15;
  const float* xr = x + (size_t)row * 1024;
  float acc = 0.f;
#pragma unroll 8
  for (int k = 0; k < 1024; ++k) acc += xr[k] * Wm[k * 16 + h];
  acc += bm[h];
  const float mval = (acc > 20.f) ? acc : log1pf(__expf(acc));

  const bf16* q = qkb + (size_t)row * 1024 + h * HDIM;
  float sacc = 0.f;
#pragma unroll 8
  for (int d = 0; d < HDIM; ++d) { const float v = (float)q[d]; sacc += v * v; }

  const int bh = (row >> 11) * HEADS + h;
  const int s  = row & 2047;
  mass[bh * SEQ + s] = mval;
  sq[bh * SEQ + s]   = sacc;
}

// ---------------------------------------------------------------------------
// Flash-style attention. 256 threads = 8 waves; wave owns 16 query rows.
// qk_j [32][64] and v_j [64][32] tiles double-buffered in LDS; step j+1's
// async-to-LDS DMA overlaps step j's WMMAs. grid = (SEQ/128, B*HEADS).
__global__ void __launch_bounds__(256)
attn_kernel(const bf16* __restrict__ qkb,   // [B*S][1024] row-major
            const bf16* __restrict__ vT,    // [(b*1024 + h*64 + d)][S]
            const float* __restrict__ mass, // [32][S]
            const float* __restrict__ sq,   // [32][S]
            float* __restrict__ out) {      // [B][S][1024]
  __shared__ __align__(16) bf16 lds_qk[2][32 * 64];  // [buf][j-local][d]  2x4 KB
  __shared__ __align__(16) bf16 lds_v[2][64 * 32];   // [buf][d][j-local]  2x4 KB
  __shared__ __align__(16) bf16 lds_p[8][16 * 32];   // per-wave p tile    8 KB

  const int tid  = threadIdx.x;
  const int wid  = tid >> 5;
  const int lane = tid & 31;
  const int l15  = lane & 15;
  const int hi   = lane >> 4;
  const int bh   = blockIdx.y;
  const int b    = bh >> 4;
  const int h    = bh & 15;
  const int i0   = blockIdx.x * 128 + wid * 16;

  // A-fragments of qk_i (16 rows x 64 d, two K=32 chunks) -- direct from global
  const bf16* qrow = qkb + (size_t)(b * SEQ + i0 + l15) * EMB + h * HDIM;
  v16bf aq[2];
#pragma unroll
  for (int c = 0; c < 2; ++c) {
    const bf16* p = qrow + c * 32 + hi * 8;
    aq[c] = load_frag_pair(p, p + 16);
  }

  // per-row stats: reg r maps to row (i0 + r + hi*8) under the C-layout
  float sqi[8], msi[8], mrun[8], lsum[8];
#pragma unroll
  for (int r = 0; r < 8; ++r) {
    const int row = i0 + r + hi * 8;
    sqi[r]  = sq[bh * SEQ + row];
    msi[r]  = mass[bh * SEQ + row];
    mrun[r] = -3.0e38f;
    lsum[r] = 0.f;
  }
  v8f acc[4];
#pragma unroll
  for (int dt = 0; dt < 4; ++dt)
    acc[dt] = (v8f){0.f, 0.f, 0.f, 0.f, 0.f, 0.f, 0.f, 0.f};

  const float* sqj_base = sq + bh * SEQ;
  const float* msj_base = mass + bh * SEQ;
  const bf16*  qj_base  = qkb + (size_t)(b * SEQ) * EMB + h * HDIM;
  const bf16*  v_base   = vT + (size_t)(b * 1024 + h * HDIM) * SEQ;
  bf16*        myp      = &lds_p[wid][0];

  // cooperative staging assignment (16B per thread per tile)
  const int qjr = tid >> 3, qseg = tid & 7;   // qk tile: 32 rows x 8 x 8 elems
  const int vdr = tid >> 2, vseg = tid & 3;   // v  tile: 64 rows x 4 x 8 elems

  auto stage_tiles = [&](int j0, int buf) {
    stage16(qj_base + (size_t)(j0 + qjr) * EMB + qseg * 8, &lds_qk[buf][qjr * 64 + qseg * 8]);
    stage16(v_base + (size_t)vdr * SEQ + j0 + vseg * 8,    &lds_v[buf][vdr * 32 + vseg * 8]);
  };

  // prologue: fill buffer 0
  stage_tiles(0, 0);
  stage_wait();
  __syncthreads();

  int cur = 0;
  for (int j0 = 0; j0 < SEQ; j0 += 32) {
    // kick off next step's DMA into the other buffer (overlaps compute below;
    // prior barrier guarantees every wave finished reading that buffer)
    if (j0 + 32 < SEQ) stage_tiles(j0 + 32, cur ^ 1);

    const bf16* qk_c = &lds_qk[cur][0];
    const bf16* v_c  = &lds_v[cur][0];

    // ---- dots = qk_i @ qk_j^T (B-frags: contiguous LDS rows) ----
    v8f sc[2];
#pragma unroll
    for (int jt = 0; jt < 2; ++jt) {
      const bf16* base = qk_c + (jt * 16 + l15) * 64;
      v16bf b0 = load_frag_c(base + 0 + hi * 8);
      v16bf b1 = load_frag_c(base + 32 + hi * 8);
      v8f z = (v8f){0.f, 0.f, 0.f, 0.f, 0.f, 0.f, 0.f, 0.f};
      z = WMMA_BF16(aq[0], b0, z);
      sc[jt] = WMMA_BF16(aq[1], b1, z);
    }

    float sqj[2], msj[2];
#pragma unroll
    for (int jt = 0; jt < 2; ++jt) {
      sqj[jt] = sqj_base[j0 + jt * 16 + l15];
      msj[jt] = msj_base[j0 + jt * 16 + l15];
    }

    // ---- scores + online softmax (fast v_rcp_f32; d2 >= 1e-6) ----
    float s[2][8], rmax[8];
#pragma unroll
    for (int r = 0; r < 8; ++r) rmax[r] = -3.0e38f;
#pragma unroll
    for (int jt = 0; jt < 2; ++jt)
#pragma unroll
      for (int r = 0; r < 8; ++r) {
        const float dot = sc[jt][r];
        const float d2  = fmaxf(sqi[r] + sqj[jt] - 2.f * dot, 0.f) + 1e-6f;
        const float sv  = msi[r] * msj[jt] * __builtin_amdgcn_rcpf(d2);
        s[jt][r] = sv;
        rmax[r]  = fmaxf(rmax[r], sv);
      }
#pragma unroll
    for (int off = 1; off < 16; off <<= 1)       // reduce within 16-lane row groups
#pragma unroll
      for (int r = 0; r < 8; ++r)
        rmax[r] = fmaxf(rmax[r], __shfl_xor(rmax[r], off, 32));

    float scale[8], psum[8];
#pragma unroll
    for (int r = 0; r < 8; ++r) {
      const float mn = fmaxf(mrun[r], rmax[r]);
      scale[r] = __expf(mrun[r] - mn);
      mrun[r]  = mn;
      const float p0 = __expf(s[0][r] - mn);
      const float p1 = __expf(s[1][r] - mn);
      s[0][r] = p0; s[1][r] = p1;
      psum[r] = p0 + p1;
    }
#pragma unroll
    for (int off = 1; off < 16; off <<= 1)
#pragma unroll
      for (int r = 0; r < 8; ++r)
        psum[r] += __shfl_xor(psum[r], off, 32);
#pragma unroll
    for (int r = 0; r < 8; ++r) lsum[r] = lsum[r] * scale[r] + psum[r];
#pragma unroll
    for (int dt = 0; dt < 4; ++dt)
#pragma unroll
      for (int r = 0; r < 8; ++r) acc[dt][r] *= scale[r];

    // ---- p: C-layout fp32 -> A-layout bf16 via per-wave LDS tile ----
#pragma unroll
    for (int jt = 0; jt < 2; ++jt)
#pragma unroll
      for (int r = 0; r < 8; ++r) {
        const int row = r + hi * 8;
        const int col = jt * 16 + l15;
        myp[row * 32 + col] = (bf16)s[jt][r];
      }
    asm volatile("s_wait_dscnt 0" ::: "memory");   // same-wave DS ordering fence
    const bf16* pr = myp + l15 * 32 + hi * 8;
    v16bf ap = load_frag_pair(pr, pr + 16);

    // ---- acc += p @ v (B-frags: contiguous LDS rows of v tile) ----
#pragma unroll
    for (int dt = 0; dt < 4; ++dt) {
      v16bf bv = load_frag_c(v_c + (dt * 16 + l15) * 32 + hi * 8);
      acc[dt] = WMMA_BF16(ap, bv, acc[dt]);
    }

    stage_wait();      // next buffer's DMA complete (this wave's asyncs)
    __syncthreads();   // all waves done reading cur + done staging
    cur ^= 1;
  }

  // ---- normalize + store ----
#pragma unroll
  for (int dt = 0; dt < 4; ++dt)
#pragma unroll
    for (int r = 0; r < 8; ++r) {
      const int row = i0 + r + hi * 8;
      const int col = h * HDIM + dt * 16 + l15;
      out[((size_t)b * SEQ + row) * EMB + col] = acc[dt][r] / lsum[r];
    }
}

// ---------------------------------------------------------------------------
extern "C" void kernel_launch(void* const* d_in, const int* in_sizes, int n_in,
                              void* d_out, int out_size, void* d_ws, size_t ws_size,
                              hipStream_t stream) {
  const float* x   = (const float*)d_in[0];
  const float* Wqk = (const float*)d_in[1];
  const float* bqk = (const float*)d_in[2];
  const float* Wm  = (const float*)d_in[3];
  const float* bm  = (const float*)d_in[4];
  const float* Wv  = (const float*)d_in[5];
  const float* bv  = (const float*)d_in[6];
  float* out = (float*)d_out;

  char* ws = (char*)d_ws;
  bf16*  xb    = (bf16*)(ws);                              // 8 MB
  bf16*  wqkT  = (bf16*)(ws + (size_t)(8u  << 20));        // 2 MB
  bf16*  wvT   = (bf16*)(ws + (size_t)(10u << 20));        // 2 MB
  bf16*  qkb   = (bf16*)(ws + (size_t)(12u << 20));        // 8 MB
  bf16*  vT    = (bf16*)(ws + (size_t)(20u << 20));        // 8 MB
  float* massA = (float*)(ws + (size_t)(28u << 20));       // 256 KB
  float* sqA   = (float*)(ws + (size_t)(28u << 20) + (256u << 10)); // 256 KB

  cvt_kernel<<<(ROWS * EMB) / 256, 256, 0, stream>>>(x, xb, ROWS * EMB);
  cvt_t_kernel<<<(EMB * EMB) / 256, 256, 0, stream>>>(Wqk, wqkT);
  cvt_t_kernel<<<(EMB * EMB) / 256, 256, 0, stream>>>(Wv, wvT);

  dim3 ggrid(ROWS / 128, EMB / 128);
  gemm_bf16_kernel<0><<<ggrid, 256, 0, stream>>>(xb, wqkT, bqk, qkb);
  gemm_bf16_kernel<1><<<ggrid, 256, 0, stream>>>(xb, wvT,  bv,  vT);

  mass_sq_kernel<<<(ROWS * HEADS) / 256, 256, 0, stream>>>(x, Wm, bm, qkb, massA, sqA);

  dim3 agrid(SEQ / 128, BSZ * HEADS);
  attn_kernel<<<agrid, 256, 0, stream>>>(qkb, vT, massA, sqA, out);
}
